// DotProductAttention_7387343749847
// MI455X (gfx1250) — matrix operand order repaired and checked by reading, not compiled, not run on hardware
//
#include <hip/hip_runtime.h>
#include <hip/hip_bf16.h>

// ---------------------------------------------------------------------------
// Fused masked+biased softmax attention for MI455X (gfx1250, wave32, WMMA).
//
// Roofline: attn output (537 MB) dominates -> ~29us floor @ 23.3 TB/s HBM.
// Matmuls use v_wmma_f32_16x16x32_f16 (inputs ~N(0,1), softmax kept fp32).
// 148 KB of LDS per workgroup (CDNA5 WGP has 320 KB) holds the full 16x2048
// score panel so softmax needs no online rescaling and attn is written once,
// non-temporally (keeps bias/mask/K/V hot in the 192 MB L2 across heads).
// ---------------------------------------------------------------------------

typedef __attribute__((ext_vector_type(16))) _Float16 v16h;
typedef __attribute__((ext_vector_type(8)))  float    v8f;
typedef __attribute__((ext_vector_type(4)))  float    vf4;
typedef __attribute__((ext_vector_type(4)))  int      vi4;

#define NB   2
#define NH   16
#define NL   2048
#define ND   64
#define NBH  (NB * NH)
#define INV_TEMP  0.125f       /* 1/sqrt(64) */
#define NEG_INF_F (-1.0e9f)

// LDS layout (dynamic): padded fp32 score panel + per-wave K^T f16 staging +
// partial-O reduction buffer.  2052 = 2048 + 4 pad (stride%64banks = 4 -> no
// 16-lane bank collisions on the strided A-fragment reads in phase 3).
#define SSTRIDE      2052
#define SCORE_BYTES  (16 * SSTRIDE * 4)          /* 131328 */
#define KT_BYTES     (8 * 64 * 16 * 2)           /*  16384 */
#define POUT_BYTES   (4 * 16 * 16 * 4)           /*   4096 */
#define SMEM_BYTES   (SCORE_BYTES + KT_BYTES + POUT_BYTES)  /* 151808 < 320KB */

__global__ __launch_bounds__(256)
void fused_attn_kernel(const float* __restrict__ qg, const float* __restrict__ kg,
                       const float* __restrict__ vg, const int* __restrict__ maskg,
                       const float* __restrict__ biasg,
                       float* __restrict__ out_o, float* __restrict__ out_attn)
{
    extern __shared__ char smem[];
    float*    scores = reinterpret_cast<float*>(smem);
    _Float16* kT     = reinterpret_cast<_Float16*>(smem + SCORE_BYTES);
    float*    pout   = reinterpret_cast<float*>(smem + SCORE_BYTES + KT_BYTES);

    const int bh = blockIdx.x >> 7;      // 128 q-tiles per (b,h)
    const int qt = blockIdx.x & 127;
    const int bb = bh >> 4;              // batch index (H = 16)
    const int q0 = qt << 4;              // first query row of this tile

    const int tid  = threadIdx.x;
    const int lane = tid & 31;
    const int wave = tid >> 5;           // 8 waves
    const int hh   = lane >> 4;          // half-wave select (ISA A/C layouts)
    const int mrow = lane & 15;          // matrix row owned by this lane
    const int ncol = lane & 15;

    const size_t head_base = (size_t)bh * NL * ND;

    // ================= Phase 1: S = Q * K^T (raw, fp32 to LDS) =============
    // A-fragments for Q (16x64 -> two 16x32 f16 frags), resident all phase.
    // ISA 16-bit A 16x32 layout: lane m(<16) holds K = {b0..b0+7, 16+b0..},
    // b0 = 8*(lane>=16).
    v16h aq0, aq1;
    {
        const float* qrow = qg + head_base + (size_t)(q0 + mrow) * ND;
        const int b0 = hh * 8;
        #pragma unroll
        for (int blk = 0; blk < 4; ++blk) {          // d-bases 0,16,32,48 (+b0)
            const float* src = qrow + blk * 16 + b0;
            vf4 f0 = *(const vf4*)(src);
            vf4 f1 = *(const vf4*)(src + 4);
            v16h* dst = (blk < 2) ? &aq0 : &aq1;
            const int e0 = (blk & 1) * 8;
            (*dst)[e0 + 0] = (_Float16)f0.x; (*dst)[e0 + 1] = (_Float16)f0.y;
            (*dst)[e0 + 2] = (_Float16)f0.z; (*dst)[e0 + 3] = (_Float16)f0.w;
            (*dst)[e0 + 4] = (_Float16)f1.x; (*dst)[e0 + 5] = (_Float16)f1.y;
            (*dst)[e0 + 6] = (_Float16)f1.z; (*dst)[e0 + 7] = (_Float16)f1.w;
        }
    }

    _Float16* kTw = kT + wave * (64 * 16);           // per-wave K^T staging
    const float* kbh = kg + head_base;

    for (int kt = wave; kt < 128; kt += 8) {         // 16 key tiles per wave
        const float* ktile = kbh + (size_t)kt * 16 * ND;   // 16 keys x 64 d
        if (kt + 8 < 128)                            // wave-uniform branch
            __builtin_prefetch(ktile + 8 * 16 * ND + lane * 32, 0, 1);

        // Stage K tile transposed as f16: kTw[d][n]. Global reads coalesced
        // (512B per wave-iteration), LDS writes are ds_store_b16.
        #pragma unroll
        for (int it = 0; it < 8; ++it) {
            const int e0 = (it * 32 + lane) * 4;     // 0..1023 step 4
            const int n  = e0 >> 6;                  // key within tile
            const int d  = e0 & 63;
            const vf4 f = *(const vf4*)(ktile + n * 64 + d);
            kTw[(d + 0) * 16 + n] = (_Float16)f.x;
            kTw[(d + 1) * 16 + n] = (_Float16)f.y;
            kTw[(d + 2) * 16 + n] = (_Float16)f.z;
            kTw[(d + 3) * 16 + n] = (_Float16)f.w;
        }
        // Wave-local LDS visibility (cross-lane transpose within one wave).
        asm volatile("s_wait_dscnt 0x0" ::: "memory");

        // B 32x16 f16 layout: lane j holds column N=j%16, K = {b0..b0+7, 16+b0..}.
        v8f cf = {};
        {
            v16h bk;
            const _Float16* src = kTw + (hh * 8) * 16 + ncol;      // d-chunk 0
            #pragma unroll
            for (int t = 0; t < 16; ++t) bk[t] = src[(t < 8 ? t : t + 8) * 16];
            cf = __builtin_amdgcn_wmma_f32_16x16x32_f16(false, aq0, false, bk,
                                                        (short)0, cf, false, false);
        }
        {
            v16h bk;
            const _Float16* src = kTw + (32 + hh * 8) * 16 + ncol; // d-chunk 1
            #pragma unroll
            for (int t = 0; t < 16; ++t) bk[t] = src[(t < 8 ? t : t + 8) * 16];
            cf = __builtin_amdgcn_wmma_f32_16x16x32_f16(false, aq1, false, bk,
                                                        (short)0, cf, false, false);
        }

        // C/D layout: VGPR r -> rows r (lanes 0-15) / r+8 (lanes 16-31).
        float* sdst = scores + kt * 16 + ncol;
        #pragma unroll
        for (int r = 0; r < 8; ++r)
            sdst[(r + hh * 8) * SSTRIDE] = cf[r];
    }
    __syncthreads();

    // ========= Phase 2: scale + bias + mask + softmax (fp32, in LDS) =======
    {
        const int row = tid >> 4;                    // 16 threads per row
        const int sub = tid & 15;
        float* srow = scores + row * SSTRIDE;
        const size_t bmrow = ((size_t)bb * NL + (q0 + row)) * (size_t)NL;
        const vf4* bias4 = (const vf4*)(biasg + bmrow);
        const vi4* mask4 = (const vi4*)(maskg + bmrow);

        float lmax = NEG_INF_F;
        #pragma unroll 4
        for (int i = 0; i < 32; ++i) {
            const int c4 = i * 16 + sub;
            vf4 s = *(vf4*)(srow + c4 * 4);
            const vf4 bvec = bias4[c4];
            const vi4 mvec = mask4[c4];
            s.x = mvec.x ? fmaf(s.x, INV_TEMP, bvec.x) : NEG_INF_F;
            s.y = mvec.y ? fmaf(s.y, INV_TEMP, bvec.y) : NEG_INF_F;
            s.z = mvec.z ? fmaf(s.z, INV_TEMP, bvec.z) : NEG_INF_F;
            s.w = mvec.w ? fmaf(s.w, INV_TEMP, bvec.w) : NEG_INF_F;
            *(vf4*)(srow + c4 * 4) = s;
            lmax = fmaxf(lmax, fmaxf(fmaxf(s.x, s.y), fmaxf(s.z, s.w)));
        }
        #pragma unroll
        for (int off = 8; off >= 1; off >>= 1)       // reduce within 16-group
            lmax = fmaxf(lmax, __shfl_xor(lmax, off, 32));

        float lsum = 0.0f;
        #pragma unroll 4
        for (int i = 0; i < 32; ++i) {
            const int c4 = i * 16 + sub;
            vf4 s = *(vf4*)(srow + c4 * 4);
            s.x = __expf(s.x - lmax); s.y = __expf(s.y - lmax);
            s.z = __expf(s.z - lmax); s.w = __expf(s.w - lmax);
            *(vf4*)(srow + c4 * 4) = s;
            lsum += s.x + s.y + s.z + s.w;
        }
        #pragma unroll
        for (int off = 8; off >= 1; off >>= 1)
            lsum += __shfl_xor(lsum, off, 32);
        const float rinv = 1.0f / lsum;              // all-masked row -> 1/L

        // Normalize in LDS and stream attn out non-temporally (537 MB total:
        // NT keeps bias/mask/K/V resident in L2 across heads/q-tiles).
        float* arow = out_attn + ((size_t)bh * NL + (q0 + row)) * (size_t)NL;
        #pragma unroll 4
        for (int i = 0; i < 32; ++i) {
            const int c4 = i * 16 + sub;
            vf4 s = *(vf4*)(srow + c4 * 4);
            s *= rinv;
            *(vf4*)(srow + c4 * 4) = s;
            __builtin_nontemporal_store(s, (vf4*)(arow + c4 * 4));
        }
    }
    __syncthreads();

    // ================= Phase 3: O = P * V (WMMA f16) =======================
    // 8 waves: n-group = wave&3 (16 of 64 out cols), k-half = wave>>2.
    {
        const int ng = wave & 3;
        const int kh = wave >> 2;
        const float* vbh = vg + head_base;
        const int b0 = hh * 8;
        v8f acc = {};
        for (int cc = 0; cc < 32; ++cc) {
            const int c = kh * 32 + cc;              // 32-key chunk index
            // A-frag from normalized P in LDS (ds_load_b128 x4, cvt to f16).
            v16h ap;
            const float* prow = scores + mrow * SSTRIDE + c * 32;
            #pragma unroll
            for (int blk = 0; blk < 2; ++blk) {      // k-offsets b0, 16+b0
                const float* src = prow + blk * 16 + b0;
                vf4 f0 = *(const vf4*)(src);
                vf4 f1 = *(const vf4*)(src + 4);
                const int e0 = blk * 8;
                ap[e0 + 0] = (_Float16)f0.x; ap[e0 + 1] = (_Float16)f0.y;
                ap[e0 + 2] = (_Float16)f0.z; ap[e0 + 3] = (_Float16)f0.w;
                ap[e0 + 4] = (_Float16)f1.x; ap[e0 + 5] = (_Float16)f1.y;
                ap[e0 + 6] = (_Float16)f1.z; ap[e0 + 7] = (_Float16)f1.w;
            }
            // B-frag: lane j holds V col (ng*16+j%16), rows c*32 + {b0..b0+7, 16+b0..}.
            v16h bfv;
            const float* vrow = vbh + (size_t)(c * 32 + b0) * ND + ng * 16 + ncol;
            #pragma unroll
            for (int t = 0; t < 16; ++t)
                bfv[t] = (_Float16)vrow[(size_t)(t < 8 ? t : t + 8) * ND];
            acc = __builtin_amdgcn_wmma_f32_16x16x32_f16(false, ap, false, bfv,
                                                         (short)0, acc, false, false);
        }

        // Pairwise cross-wave reduction: waves 4-7 dump partials to LDS.
        if (wave >= 4) {
            float* pdst = pout + (wave - 4) * 256;
            #pragma unroll
            for (int r = 0; r < 8; ++r) pdst[r * 32 + lane] = acc[r];
        }
        __syncthreads();
        if (wave < 4) {
            const float* psrc = pout + wave * 256;
            float* orow = out_o + head_base + (size_t)q0 * ND + ng * 16 + ncol;
            #pragma unroll
            for (int r = 0; r < 8; ++r) {
                const float val = acc[r] + psrc[r * 32 + lane];
                orow[(size_t)(r + hh * 8) * ND] = val;
            }
        }
    }
}

extern "C" void kernel_launch(void* const* d_in, const int* in_sizes, int n_in,
                              void* d_out, int out_size, void* d_ws, size_t ws_size,
                              hipStream_t stream) {
    (void)in_sizes; (void)n_in; (void)out_size; (void)d_ws; (void)ws_size;
    const float* q    = (const float*)d_in[0];
    const float* k    = (const float*)d_in[1];
    const float* v    = (const float*)d_in[2];
    const int*   mask = (const int*)d_in[3];
    const float* bias = (const float*)d_in[4];

    float* out_o    = (float*)d_out;                              // [B,H,L,D]
    float* out_attn = out_o + (size_t)NB * NH * NL * ND;          // [B,H,L,L]

    // 148 KB dynamic LDS (> default cap; gfx1250 WGP supports 320 KB).
    (void)hipFuncSetAttribute((const void*)fused_attn_kernel,
                              hipFuncAttributeMaxDynamicSharedMemorySize,
                              SMEM_BYTES);

    fused_attn_kernel<<<dim3(NBH * (NL / 16)), dim3(256), SMEM_BYTES, stream>>>(
        q, k, v, mask, bias, out_o, out_attn);
}
